// SFC_GATLayer_59622736003360
// MI455X (gfx1250) — matrix operand order repaired and checked by reading, not compiled, not run on hardware
//
#include <hip/hip_runtime.h>
#include <math.h>

// ---------------- problem constants (from reference) ----------------
#define GN     8192
#define IN_F   256
#define OUT_F  128
#define GE     262144
#define GALPHA 0.2f
#define GNEG   -9.0e15f
#define BK     32          // K-chunk = one bf16 WMMA K dimension

typedef __attribute__((ext_vector_type(16))) __bf16 v16bf;
typedef __attribute__((ext_vector_type(2)))  __bf16 v2bf;
typedef __attribute__((ext_vector_type(8)))  float  v8f;

union BFV { v16bf v; unsigned u[8]; };
union BF2 { v2bf v; unsigned u; };

__device__ __forceinline__ unsigned bf16b(float x) {
  unsigned u = __float_as_uint(x);
  return (u + 0x7fffu + ((u >> 16) & 1u)) >> 16;   // RNE f32->bf16
}
__device__ __forceinline__ unsigned pk2(float lo, float hi) {
#if __has_builtin(__builtin_amdgcn_cvt_pk_bf16_f32)
  BF2 r;
  r.v = __builtin_amdgcn_cvt_pk_bf16_f32(lo, hi);  // single v_cvt_pk_bf16_f32
  return r.u;
#else
  return bf16b(lo) | (bf16b(hi) << 16);
#endif
}

// ---------------- helper kernels ----------------
__global__ void zero_u4(uint4* __restrict__ p, int n4) {
  int i = blockIdx.x * blockDim.x + threadIdx.x;
  if (i < n4) p[i] = make_uint4(0u, 0u, 0u, 0u);
}

// edge list -> 1-bit-per-entry adjacency mask [GN][GN/32]
// (JAX default x64-disabled => edge_index arrives as int32)
__global__ void scatter_edges(const int* __restrict__ ei, unsigned* __restrict__ adj) {
  int t = blockIdx.x * blockDim.x + threadIdx.x;
  if (t < GE) {
    int r = ei[t];
    int c = ei[GE + t];
    atomicOr(&adj[(size_t)r * (GN / 32) + (c >> 5)], 1u << (c & 31));
  }
}

// Wh = h @ W  (f32, tiny) ; also emit transposed bf16 copy for the WMMA B operand
__global__ __launch_bounds__(OUT_F) void wh_kernel(const float* __restrict__ h,
                                                   const float* __restrict__ W,
                                                   float* __restrict__ Wh,
                                                   unsigned short* __restrict__ WhT) {
  int i = blockIdx.x;        // row
  int f = threadIdx.x;       // output feature
  const float* hr = h + (size_t)i * IN_F;
  float acc = 0.f;
#pragma unroll 8
  for (int k = 0; k < IN_F; ++k) acc = fmaf(hr[k], W[(size_t)k * OUT_F + f], acc);
  Wh[(size_t)i * OUT_F + f] = acc;
  WhT[(size_t)f * GN + i]   = (unsigned short)bf16b(acc);   // [OUT_F][GN] bf16
}

// Wh1 = Wh @ a[:128], Wh2 = Wh @ a[128:]
__global__ void wh12_kernel(const float* __restrict__ Wh, const float* __restrict__ a,
                            float* __restrict__ Wh1, float* __restrict__ Wh2) {
  int i = blockIdx.x * blockDim.x + threadIdx.x;
  if (i >= GN) return;
  const float* r = Wh + (size_t)i * OUT_F;
  float s1 = 0.f, s2 = 0.f;
#pragma unroll 8
  for (int f = 0; f < OUT_F; ++f) {
    float v = r[f];
    s1 = fmaf(v, a[f], s1);
    s2 = fmaf(v, a[OUT_F + f], s2);
  }
  Wh1[i] = s1;
  Wh2[i] = s2;
}

// ---------------- fused flash-style GAT kernel (K-split) ----------------
// blockIdx.x = 16-row block, blockIdx.y = K-split index. One wave per block.
// 8 C tiles (16x16 f32) cover all 128 output features.
// Lane L: softmax row = L&15, khalf = L>>4.
// A operand (bf16 16x32, ISA 7.12.2): lane<16 owns K {0..7,16..23}, lane>=16 owns {8..15,24..31}.
// B operand (bf16 32x16): lanes 0-15 col N=L, K {0..7,16..23}; lanes 16-31 col N=L-16, K {8..15,24..31}
//   -> two 8-bf16 runs along K from WhT => two uint4 loads per tile per lane.
__global__ __launch_bounds__(32) void gat_fused(const float* __restrict__ S,
                                                const unsigned* __restrict__ adj,
                                                const unsigned short* __restrict__ WhT,
                                                const float* __restrict__ Wh1,
                                                const float* __restrict__ Wh2,
                                                const float* __restrict__ rg,
                                                float* __restrict__ accP,   // [split][GN][OUT_F]
                                                float* __restrict__ mP,     // [split][GN]
                                                float* __restrict__ lP,     // [split][GN]
                                                int splitLen) {
  const int lane  = threadIdx.x;
  const int row   = lane & 15;
  const int khalf = lane >> 4;
  const int i0    = blockIdx.x * 16;
  const int sidx  = blockIdx.y;
  const int jb    = sidx * splitLen;
  const int je    = jb + splitLen;

  const float gamma = log1pf(__expf(rg[0]));     // softplus(raw_gamma)
  const float wh1   = Wh1[i0 + row];

  float m = -3.0e38f;   // running row max (lanes row / row+16 hold same value)
  float l = 0.0f;       // running row sum
  v8f acc[8];
#pragma unroll
  for (int t = 0; t < 8; ++t) acc[t] = (v8f){0.f, 0.f, 0.f, 0.f, 0.f, 0.f, 0.f, 0.f};

  const float*    Srow   = S   + (size_t)(i0 + row) * GN;
  const unsigned* adjrow = adj + (size_t)(i0 + row) * (GN / 32);
  const int kb8  = khalf * 8;    // A-layout sub-base
  const int kb16 = khalf * 16;   // unused (B loads also use kb8)

  for (int j0 = jb; j0 < je; j0 += BK) {
    // prefetch the HBM-resident S stream 4KB ahead (speculative, OOB drops silently)
    __builtin_prefetch(Srow + j0 + 1024, 0, 0);

    // ---- loads for this chunk (coalesced across the wave) ----
    unsigned bits = adjrow[j0 >> 5];
    float sa[16], wa[16];
    *(float4*)&sa[0]  = *(const float4*)(Srow + j0 + kb8);
    *(float4*)&sa[4]  = *(const float4*)(Srow + j0 + kb8 + 4);
    *(float4*)&sa[8]  = *(const float4*)(Srow + j0 + kb8 + 16);
    *(float4*)&sa[12] = *(const float4*)(Srow + j0 + kb8 + 20);
    *(float4*)&wa[0]  = *(const float4*)(Wh2 + j0 + kb8);
    *(float4*)&wa[4]  = *(const float4*)(Wh2 + j0 + kb8 + 4);
    *(float4*)&wa[8]  = *(const float4*)(Wh2 + j0 + kb8 + 16);
    *(float4*)&wa[12] = *(const float4*)(Wh2 + j0 + kb8 + 20);

    // ---- attention scores in A-operand register order ----
    float att[16];
    float cmax = -3.0e38f;
#pragma unroll
    for (int t = 0; t < 16; ++t) {
      int k  = ((t & 8) << 1) + kb8 + (t & 7);       // K index within chunk
      float e = wh1 + wa[t];
      e = e > 0.f ? e : GALPHA * e;                  // leaky_relu
      float v = ((bits >> k) & 1u) ? e : GNEG;       // adjacency mask
      v *= fmaf(gamma, sa[t], 1.0f);                 // * (1 + gamma*S)
      att[t] = v;
      cmax = fmaxf(cmax, v);
    }
    cmax = fmaxf(cmax, __shfl_xor(cmax, 16));        // combine row halves

    // ---- online softmax update ----
    float mold = m;
    m = fmaxf(m, cmax);
#if __has_builtin(__builtin_amdgcn_ballot_w32)
    bool need_rescale = __builtin_amdgcn_ballot_w32(m > mold) != 0u;
#else
    bool need_rescale = true;
#endif
    if (need_rescale) {
      float corr = __expf(mold - m);
      l *= corr;
#pragma unroll
      for (int r = 0; r < 8; ++r) {
        float fc = __shfl(corr, r + (khalf << 3));   // factor for this VGPR's row
#pragma unroll
        for (int t = 0; t < 8; ++t) acc[t][r] *= fc;
      }
    }

    BFV A;
    float psum = 0.f;
#pragma unroll
    for (int t = 0; t < 16; t += 2) {
      float p0 = __expf(att[t] - m);
      float p1 = __expf(att[t + 1] - m);
      psum += p0 + p1;
      A.u[t >> 1] = pk2(p0, p1);                     // v_cvt_pk_bf16_f32
    }
    psum += __shfl_xor(psum, 16);
    l += psum;

    // ---- P(bf16) @ Wh(bf16) : 8 WMMA tiles cover all 128 features ----
    const unsigned short* bbase = WhT + j0 + kb8;
#pragma unroll
    for (int t = 0; t < 8; ++t) {
      int f = t * 16 + row;                          // this lane's B column
      const uint4* bp = (const uint4*)(bbase + (size_t)f * GN);
      uint4 b0 = bp[0];
      uint4 b1 = bp[2];
      BFV B;
      B.u[0] = b0.x; B.u[1] = b0.y; B.u[2] = b0.z; B.u[3] = b0.w;
      B.u[4] = b1.x; B.u[5] = b1.y; B.u[6] = b1.z; B.u[7] = b1.w;
      acc[t] = __builtin_amdgcn_wmma_f32_16x16x32_bf16(
          /*neg_a=*/false, A.v, /*neg_b=*/false, B.v,
          /*c_mod=*/(short)0, acc[t], /*reuse_a=*/false, /*reuse_b=*/false);
    }
  }

  // ---- write partial accumulators + per-row stats (merge kernel finishes) ----
  float* accO = accP + (size_t)sidx * GN * OUT_F;
#pragma unroll
  for (int r = 0; r < 8; ++r) {
    int orow = i0 + r + khalf * 8;                   // C layout: lanes>=16 hold M=r+8
#pragma unroll
    for (int t = 0; t < 8; ++t)
      accO[(size_t)orow * OUT_F + t * 16 + (lane & 15)] = acc[t][r];
  }
  if (lane < 16) {
    mP[(size_t)sidx * GN + i0 + row] = m;
    lP[(size_t)sidx * GN + i0 + row] = l;
  }
}

// ---------------- merge splits, normalize, ELU ----------------
__global__ __launch_bounds__(256) void merge_out(const float* __restrict__ accP,
                                                 const float* __restrict__ mP,
                                                 const float* __restrict__ lP,
                                                 int split, float* __restrict__ out) {
  int idx = blockIdx.x * 256 + threadIdx.x;   // i*OUT_F + f
  if (idx >= GN * OUT_F) return;
  int i = idx >> 7;
  float M = -3.0e38f;
  for (int s = 0; s < split; ++s) M = fmaxf(M, mP[(size_t)s * GN + i]);
  float num = 0.f, den = 0.f;
  for (int s = 0; s < split; ++s) {
    float w = __expf(mP[(size_t)s * GN + i] - M);
    den = fmaf(lP[(size_t)s * GN + i], w, den);
    num = fmaf(accP[(size_t)s * GN * OUT_F + idx], w, num);
  }
  float v = num / den;
  out[idx] = v > 0.f ? v : (__expf(v) - 1.0f);     // elu
}

// ---------------- launch ----------------
extern "C" void kernel_launch(void* const* d_in, const int* in_sizes, int n_in,
                              void* d_out, int out_size, void* d_ws, size_t ws_size,
                              hipStream_t stream) {
  const float* h  = (const float*)d_in[0];
  const int*   ei = (const int*)d_in[1];     // edge_index (int32, 2*E)
  const float* S  = (const float*)d_in[2];
  const float* W  = (const float*)d_in[3];
  const float* a  = (const float*)d_in[4];
  const float* rg = (const float*)d_in[5];
  float* out = (float*)d_out;

  const size_t MB = 1024u * 1024u;
  char* ws = (char*)d_ws;
  unsigned*       adj = (unsigned*)ws;                                  // 8 MB
  float*          Wh  = (float*)(ws + 8 * MB);                          // 4 MB
  unsigned short* WhT = (unsigned short*)(ws + 12 * MB);                // 2 MB
  float*          Wh1 = (float*)(ws + 14 * MB);                         // 32 KB
  float*          Wh2 = (float*)(ws + 14 * MB + 64 * 1024);             // 32 KB
  float*          mP  = (float*)(ws + 14 * MB + 128 * 1024);            // <=128 KB
  float*          lP  = (float*)(ws + 14 * MB + 256 * 1024);            // <=128 KB
  float*          accP = (float*)(ws + 15 * MB);                        // split*4 MB

  // pick largest K-split that fits in the workspace (deterministic per run)
  int split = 4;
  if (ws_size != 0) {
    while (split > 1 && (15 * MB + (size_t)split * 4 * MB) > ws_size) split >>= 1;
  }
  const int splitLen = GN / split;

  const int adjWords = GN * (GN / 32);     // 2,097,152 u32
  zero_u4<<<adjWords / 4 / 256, 256, 0, stream>>>((uint4*)adj, adjWords / 4);
  scatter_edges<<<GE / 256, 256, 0, stream>>>(ei, adj);
  wh_kernel<<<GN, OUT_F, 0, stream>>>(h, W, Wh, WhT);
  wh12_kernel<<<GN / 256, 256, 0, stream>>>(Wh, a, Wh1, Wh2);
  gat_fused<<<dim3(GN / 16, split), 32, 0, stream>>>(S, adj, WhT, Wh1, Wh2, rg,
                                                     accP, mP, lP, splitLen);
  merge_out<<<(GN * OUT_F) / 256, 256, 0, stream>>>(accP, mP, lP, split, out);
}
